// AttentionCNN_11656541241542
// MI455X (gfx1250) — matrix operand order repaired and checked by reading, not compiled, not run on hardware
//
#include <hip/hip_runtime.h>

// ---------------------------------------------------------------------------
// AttentionCNN for MI455X (gfx1250, wave32, WMMA + TDM).
// Pipeline:
//   stage1:  dwconv3x3 + BN -> bf16 activations [M][K] (K=channels contig)
//   gemms :  bf16 WMMA (v_wmma_f32_16x16x32_bf16), f32 accumulate
//            Q [784x512], K [224x512] (padded seq), V^T [512x224], R f32+BN
//   attn  :  fused per (i-tile, head, batch) wave:
//            TDM async-loads the 64x224 V block to LDS while QK^T + softmax
//            run, then attn*V reads B tiles from LDS (s_wait_tensorcnt).
//   pool  :  deterministic reduction of per-tile partial sums
// ---------------------------------------------------------------------------

#define EPS_BN 1e-5f
#define EPS_LN 1e-5f

typedef __attribute__((ext_vector_type(8)))  __bf16 v8bf;
typedef __attribute__((ext_vector_type(16))) __bf16 v16bf;
typedef __attribute__((ext_vector_type(8)))  float  v8f;

#ifndef __has_builtin
#define __has_builtin(x) 0
#endif

#if __has_builtin(__builtin_amdgcn_tensor_load_to_lds) && \
    __has_builtin(__builtin_amdgcn_s_wait_tensorcnt)
#define USE_TDM 1
#if __has_include(<hip/amd_detail/amd_gfx1250_TDM.h>)
#define TDM_6ARG 1   // therock/clang-23 lane: 6-arg builtin
#else
#define TDM_6ARG 0   // ROCm 7.2 / clang-22 lane: 5-arg builtin
#endif
#else
#define USE_TDM 0
#endif

__device__ __forceinline__ v16bf cat16(v8bf lo, v8bf hi) {
  return __builtin_shufflevector(lo, hi, 0,1,2,3,4,5,6,7,8,9,10,11,12,13,14,15);
}

__device__ __forceinline__ v8f wmma_bf16f32(v16bf a, v16bf b, v8f c) {
  // (neg_a, A, neg_b, B, c_mod, C, reuse_a, reuse_b)
  return __builtin_amdgcn_wmma_f32_16x16x32_bf16(false, a, false, b, (short)0, c,
                                                 false, false);
}

// 16x32 bf16 A tile from row-major [*, lda]; base points at (row m0, col k0).
// ISA layout: lanes 0-15 hold M=lane, K={ks..ks+7} and {16+ks..16+ks+7},
// ks = 0 (lanes<16) or 8 (lanes>=16).
__device__ __forceinline__ v16bf load_a32(const __bf16* base, int lane, int lda) {
  int m  = lane & 15;
  int ks = (lane < 16) ? 0 : 8;
  const __bf16* p = base + (size_t)m * lda + ks;
  v8bf lo = *(const v8bf*)(p);
  v8bf hi = *(const v8bf*)(p + 16);
  return cat16(lo, hi);
}

// 32x16 bf16 B tile where the contraction dim (K) is contiguous per column.
// pcol points at (K=k0, N=this lane's column). K={ks..ks+7} and {16+ks..16+ks+7},
// ks = 0 (lanes<16) or 8 (lanes>=16).
__device__ __forceinline__ v16bf load_b32(const __bf16* pcol, int lane) {
  int ks = (lane < 16) ? 0 : 8;
  const __bf16* p = pcol + ks;
  v8bf lo = *(const v8bf*)(p);
  v8bf hi = *(const v8bf*)(p + 16);
  return cat16(lo, hi);
}

#if USE_TDM
typedef unsigned int uint32x4 __attribute__((ext_vector_type(4)));
typedef int          int32x8  __attribute__((ext_vector_type(8)));
typedef int          int32x4  __attribute__((ext_vector_type(4)));

// Issue a TDM 2D tile load (bf16 elements) Global -> LDS.
// D# layout per CDNA5 ISA ch.8 (group0: count/lds_addr/global_addr/type=2;
// group1: data_size=2B, tensor dims, tile dims, dim0 stride).
__device__ __forceinline__ void tdm_load_2d_bf16(unsigned lds_off,
                                                 unsigned long long gaddr,
                                                 int width_elems, int rows,
                                                 int row_stride_elems) {
  uint32x4 g0;
  g0[0] = 1u;                                           // count=1, user desc
  g0[1] = lds_off;                                      // lds_addr (bytes)
  g0[2] = (unsigned)(gaddr & 0xffffffffull);            // global_addr[31:0]
  g0[3] = (unsigned)((gaddr >> 32) & 0x1ffffffull)      // global_addr[56:32]
        | (2u << 30);                                   // type = 2 ("image")
  int32x8 g1;
  g1[0] = (int)(1u << 16);                              // data_size=1 -> 2 bytes
  g1[1] = (int)(((unsigned)width_elems & 0xffffu) << 16);   // tensor_dim0[15:0]
  g1[2] = (int)(((unsigned)rows & 0xffffu) << 16);          // tensor_dim1[15:0]
  g1[3] = (int)(((unsigned)width_elems & 0xffffu) << 16);   // tile_dim0
  g1[4] = rows;                                         // tile_dim1 (tile_dim2=0)
  g1[5] = row_stride_elems;                             // tensor_dim0_stride[31:0]
  g1[6] = 0;
  g1[7] = 0;
  int32x4 z4 = {0, 0, 0, 0};
#if TDM_6ARG
  int32x8 z8 = {0, 0, 0, 0, 0, 0, 0, 0};
  __builtin_amdgcn_tensor_load_to_lds(g0, g1, z4, z4, z8, 0);
#else
  __builtin_amdgcn_tensor_load_to_lds(g0, g1, z4, z4, 0);
#endif
}
#endif  // USE_TDM

// ---------------------------------------------------------------------------
// Elementwise stage-1 kernels
// ---------------------------------------------------------------------------

__global__ void cvt_bf16(const float* __restrict__ in, __bf16* __restrict__ out, int n) {
  int i = blockIdx.x * 256 + threadIdx.x;
  if (i < n) out[i] = (__bf16)in[i];
}

// residual 1x1 conv weights with BN folded: w' = w * g/sqrt(v+eps), beta = b - m*inv
__global__ void cvt_wds(const float* __restrict__ w, const float* __restrict__ g,
                        const float* __restrict__ bb, const float* __restrict__ m,
                        const float* __restrict__ v, __bf16* __restrict__ out,
                        float* __restrict__ beta) {
  int i = blockIdx.x * 256 + threadIdx.x;
  if (i >= 64 * 640) return;
  int d = i / 640;
  float inv = g[d] * rsqrtf(v[d] + EPS_BN);
  out[i] = (__bf16)(w[i] * inv);
  if ((i % 640) == 0) beta[d] = bb[d] - m[d] * inv;
}

// qact[n][hw][c] = BN(dwconv3x3 stride1(x)), xact[n][hw][c] = bf16(x) (for residual GEMM)
__global__ void stage1_q(const float* __restrict__ x, const float* __restrict__ w,
                         const float* __restrict__ g, const float* __restrict__ bb,
                         const float* __restrict__ m, const float* __restrict__ v,
                         __bf16* __restrict__ qact, __bf16* __restrict__ xact) {
  int n = blockIdx.y;
  int idx = blockIdx.x * 256 + threadIdx.x;
  if (idx >= 784 * 640) return;
  int hw = idx % 784;
  int c  = idx / 784;
  int oh = hw / 28, ow = hw % 28;
  const float* xc = x + ((size_t)n * 640 + c) * 784;
  float acc = 0.f;
  #pragma unroll
  for (int ky = 0; ky < 3; ++ky) {
    int ih = oh - 1 + ky;
    if (ih < 0 || ih >= 28) continue;
    #pragma unroll
    for (int kx = 0; kx < 3; ++kx) {
      int iw = ow - 1 + kx;
      if (iw < 0 || iw >= 28) continue;
      acc += xc[ih * 28 + iw] * w[c * 9 + ky * 3 + kx];
    }
  }
  float inv = g[c] * rsqrtf(v[c] + EPS_BN);
  float val = (acc - m[c]) * inv + bb[c];
  size_t o = ((size_t)n * 784 + hw) * 640 + c;
  qact[o] = (__bf16)val;
  xact[o] = (__bf16)xc[hw];
}

// kvact[n][j][c] for j in [0,224): BN(dwconv3x3 stride2(x)) for j<196, else 0 (seq pad)
__global__ void stage1_kv(const float* __restrict__ x, const float* __restrict__ w,
                          const float* __restrict__ g, const float* __restrict__ bb,
                          const float* __restrict__ m, const float* __restrict__ v,
                          __bf16* __restrict__ kvact) {
  int n = blockIdx.y;
  int idx = blockIdx.x * 256 + threadIdx.x;
  if (idx >= 224 * 640) return;
  int j = idx % 224;
  int c = idx / 224;
  float val = 0.f;
  if (j < 196) {
    int oh = j / 14, ow = j % 14;
    const float* xc = x + ((size_t)n * 640 + c) * 784;
    float acc = 0.f;
    #pragma unroll
    for (int ky = 0; ky < 3; ++ky) {
      int ih = 2 * oh - 1 + ky;
      if (ih < 0 || ih >= 28) continue;
      #pragma unroll
      for (int kx = 0; kx < 3; ++kx) {
        int iw = 2 * ow - 1 + kx;
        if (iw < 0 || iw >= 28) continue;
        acc += xc[ih * 28 + iw] * w[c * 9 + ky * 3 + kx];
      }
    }
    float inv = g[c] * rsqrtf(v[c] + EPS_BN);
    val = (acc - m[c]) * inv + bb[c];
  }
  kvact[((size_t)n * 224 + j) * 640 + c] = (__bf16)val;
}

// ---------------------------------------------------------------------------
// Generic single-wave WMMA GEMM: C[16x64 tile] = A[M][K] * Bt[N][K]^T
// mode 0: bf16 row-major C (ldc)   mode 1: bf16 transposed C[n][m] (ldc)
// mode 2: f32 row-major C + bias[n]
// ---------------------------------------------------------------------------
__global__ __launch_bounds__(32)
void gemm_bf16(const __bf16* __restrict__ A, const __bf16* __restrict__ Bt,
               void* __restrict__ Cout, const float* __restrict__ bias,
               int K, long long strideA, long long strideC, int ldc, int mode) {
  int lane = threadIdx.x;
  int m0 = blockIdx.x * 16;
  int n0 = blockIdx.y * 64;
  int b  = blockIdx.z;
  const __bf16* Ab = A + (size_t)b * strideA;

  v8f acc[4];
  #pragma unroll
  for (int t = 0; t < 4; ++t)
    #pragma unroll
    for (int i = 0; i < 8; ++i) acc[t][i] = 0.0f;

  int ncol = lane & 15;
  for (int kc = 0; kc < K; kc += 32) {
    v16bf a = load_a32(Ab + (size_t)m0 * K + kc, lane, K);
    #pragma unroll
    for (int t = 0; t < 4; ++t) {
      const __bf16* pcol = Bt + (size_t)(n0 + t * 16 + ncol) * K + kc;
      v16bf bt = load_b32(pcol, lane);
      acc[t] = wmma_bf16f32(a, bt, acc[t]);
    }
  }

  int rowh = (lane >> 4) * 8;  // C/D layout: M = r + 8*(lane>=16), N = lane&15
  if (mode == 0) {
    __bf16* C = (__bf16*)Cout + (size_t)b * strideC;
    #pragma unroll
    for (int t = 0; t < 4; ++t)
      #pragma unroll
      for (int r = 0; r < 8; ++r)
        C[(size_t)(m0 + rowh + r) * ldc + (n0 + t * 16 + ncol)] = (__bf16)acc[t][r];
  } else if (mode == 1) {
    __bf16* C = (__bf16*)Cout + (size_t)b * strideC;
    #pragma unroll
    for (int t = 0; t < 4; ++t)
      #pragma unroll
      for (int r = 0; r < 8; ++r)
        C[(size_t)(n0 + t * 16 + ncol) * ldc + (m0 + rowh + r)] = (__bf16)acc[t][r];
  } else {
    float* C = (float*)Cout + (size_t)b * strideC;
    #pragma unroll
    for (int t = 0; t < 4; ++t)
      #pragma unroll
      for (int r = 0; r < 8; ++r) {
        int cc = n0 + t * 16 + ncol;
        C[(size_t)(m0 + rowh + r) * ldc + cc] = acc[t][r] + bias[cc];
      }
  }
}

// ---------------------------------------------------------------------------
// Fused attention: one wave per (i-tile of 16, head, batch).
// Q[n][784][512] bf16, Kb[n][224][512] bf16 (rows>=196 zero),
// Vt[n][512][224] bf16, R[n][784][64] f32 (residual with BN folded),
// partials[n][h][49][64] f32 (sum over 16 rows of LN output).
// The 64x224 V block is TDM-async-loaded to LDS during QK^T + softmax.
// ---------------------------------------------------------------------------
__global__ __launch_bounds__(32)
void attn_fused(const __bf16* __restrict__ Q, const __bf16* __restrict__ Kb,
                const __bf16* __restrict__ Vt, const float* __restrict__ R,
                const float* __restrict__ ln_g, const float* __restrict__ ln_b,
                float* __restrict__ partials) {
  __shared__ __align__(32)  __bf16 sattn[16 * 224];
#if USE_TDM
  __shared__ __align__(128) __bf16 svt[64 * 224];
#endif
  int lane = threadIdx.x;
  int it = blockIdx.x, h = blockIdx.y, n = blockIdx.z;
  int i0 = it * 16;
  int ncol = lane & 15;
  int rowh = (lane >> 4) * 8;

  const __bf16* vtbase = Vt + (size_t)n * 512 * 224 + (size_t)(h * 64) * 224;
#if USE_TDM
  // Kick off the async V-block DMA; it overlaps with QK^T and the softmax.
  tdm_load_2d_bf16((unsigned)(size_t)(void*)&svt[0],
                   (unsigned long long)(size_t)(const void*)vtbase,
                   /*width*/ 224, /*rows*/ 64, /*row stride*/ 224);
#endif

  // ---- dots = (q k^T) * scale : A = q[16x64], B = k^T (K=d contiguous) ----
  const __bf16* qbase = Q + ((size_t)n * 784 + i0) * 512 + h * 64;
  v16bf aq0 = load_a32(qbase,      lane, 512);
  v16bf aq1 = load_a32(qbase + 32, lane, 512);

  v8f dots[14];
  #pragma unroll
  for (int jt = 0; jt < 14; ++jt)
    #pragma unroll
    for (int i = 0; i < 8; ++i) dots[jt][i] = 0.f;

  const __bf16* kbase = Kb + (size_t)n * 224 * 512 + h * 64;
  #pragma unroll
  for (int jt = 0; jt < 14; ++jt) {
    const __bf16* kcol = kbase + (size_t)(jt * 16 + ncol) * 512;
    v16bf b0 = load_b32(kcol,      lane);
    v16bf b1 = load_b32(kcol + 32, lane);
    v8f d = dots[jt];
    d = wmma_bf16f32(aq0, b0, d);
    d = wmma_bf16f32(aq1, b1, d);
    dots[jt] = d;
  }

  // ---- masked row softmax (rows split across lane halves; 16-lane shuffles) ----
  #pragma unroll
  for (int r = 0; r < 8; ++r) {
    float mx = -1e30f;
    #pragma unroll
    for (int jt = 0; jt < 14; ++jt) {
      float val = dots[jt][r] * 0.125f;  // SCALE = 64^-0.5
      dots[jt][r] = val;
      if (jt * 16 + ncol < 196) mx = fmaxf(mx, val);
    }
    #pragma unroll
    for (int off = 1; off < 16; off <<= 1) mx = fmaxf(mx, __shfl_xor(mx, off, 16));
    float s = 0.f;
    #pragma unroll
    for (int jt = 0; jt < 14; ++jt) {
      float e = (jt * 16 + ncol < 196) ? __expf(dots[jt][r] - mx) : 0.f;
      dots[jt][r] = e;
      s += e;
    }
    #pragma unroll
    for (int off = 1; off < 16; off <<= 1) s += __shfl_xor(s, off, 16);
    float inv = 1.f / s;
    #pragma unroll
    for (int jt = 0; jt < 14; ++jt) dots[jt][r] *= inv;
  }

  // ---- attn: C-layout regs -> LDS [16][224] bf16 -> re-read in A layout ----
  #pragma unroll
  for (int jt = 0; jt < 14; ++jt)
    #pragma unroll
    for (int r = 0; r < 8; ++r)
      sattn[(size_t)(rowh + r) * 224 + jt * 16 + ncol] = (__bf16)dots[jt][r];
  __syncthreads();

#if USE_TDM
  __builtin_amdgcn_s_wait_tensorcnt(0);  // V block resident in LDS
#endif

  // ---- out = attn * v : B = Vt rows (K=j contiguous) ----
  v8f acc[4];
  #pragma unroll
  for (int t = 0; t < 4; ++t)
    #pragma unroll
    for (int i = 0; i < 8; ++i) acc[t][i] = 0.f;

  #pragma unroll
  for (int kc = 0; kc < 7; ++kc) {
    v16bf aA = load_a32(&sattn[kc * 32], lane, 224);  // ds_load path
    #pragma unroll
    for (int t = 0; t < 4; ++t) {
#if USE_TDM
      const __bf16* vp = &svt[(size_t)(t * 16 + ncol) * 224 + kc * 32];
#else
      const __bf16* vp = vtbase + (size_t)(t * 16 + ncol) * 224 + kc * 32;
#endif
      v16bf bV = load_b32(vp, lane);
      acc[t] = wmma_bf16f32(aA, bV, acc[t]);
    }
  }

  // ---- residual add + LayerNorm over d (64 values per row) ----
  float vals[4][8];
  const float* Rb = R + ((size_t)n * 784) * 64;
  #pragma unroll
  for (int t = 0; t < 4; ++t)
    #pragma unroll
    for (int r = 0; r < 8; ++r) {
      int i = i0 + rowh + r;
      int d = t * 16 + ncol;
      vals[t][r] = acc[t][r] + Rb[(size_t)i * 64 + d];
    }

  #pragma unroll
  for (int r = 0; r < 8; ++r) {
    float s1 = 0.f, s2 = 0.f;
    #pragma unroll
    for (int t = 0; t < 4; ++t) { s1 += vals[t][r]; s2 += vals[t][r] * vals[t][r]; }
    #pragma unroll
    for (int off = 1; off < 16; off <<= 1) {
      s1 += __shfl_xor(s1, off, 16);
      s2 += __shfl_xor(s2, off, 16);
    }
    float mean = s1 * (1.f / 64.f);
    float var  = s2 * (1.f / 64.f) - mean * mean;
    float stdv = sqrtf(fmaxf(var, 0.f));
    float rinv = 1.f / (stdv + EPS_LN);
    #pragma unroll
    for (int t = 0; t < 4; ++t) {
      int d = t * 16 + ncol;
      vals[t][r] = (vals[t][r] - mean) * rinv * ln_g[h * 64 + d] + ln_b[h * 64 + d];
    }
  }

  // ---- partial spatial pool: sum over the 16 rows of this tile, per d ----
  #pragma unroll
  for (int t = 0; t < 4; ++t) {
    float ps = 0.f;
    #pragma unroll
    for (int r = 0; r < 8; ++r) ps += vals[t][r];
    ps += __shfl_xor(ps, 16, 32);  // combine row halves (same d in both halves)
    if (lane < 16)
      partials[((((size_t)n * 8 + h) * 49) + it) * 64 + t * 16 + lane] = ps;
  }
}

__global__ void pool_final(const float* __restrict__ part, float* __restrict__ out) {
  int idx = blockIdx.x * 256 + threadIdx.x;
  if (idx >= 64 * 8 * 64) return;
  int d = idx & 63;
  int nh = idx >> 6;
  float s = 0.f;
  for (int it = 0; it < 49; ++it) s += part[((size_t)nh * 49 + it) * 64 + d];
  out[idx] = s * (1.0f / 784.0f);
}

// ---------------------------------------------------------------------------
extern "C" void kernel_launch(void* const* d_in, const int* in_sizes, int n_in,
                              void* d_out, int out_size, void* d_ws, size_t ws_size,
                              hipStream_t stream) {
  (void)in_sizes; (void)n_in; (void)out_size; (void)ws_size;
  const float* x      = (const float*)d_in[0];
  const float* w_dwq  = (const float*)d_in[1];
  const float* w_pwq  = (const float*)d_in[2];
  const float* w_dwkv = (const float*)d_in[3];
  const float* w_pwkv = (const float*)d_in[4];
  const float* w_ds   = (const float*)d_in[5];
  const float* ln_g   = (const float*)d_in[6];
  const float* ln_b   = (const float*)d_in[7];
  const float* bnq_g  = (const float*)d_in[8];
  const float* bnq_b  = (const float*)d_in[9];
  const float* bnq_m  = (const float*)d_in[10];
  const float* bnq_v  = (const float*)d_in[11];
  const float* bnkv_g = (const float*)d_in[12];
  const float* bnkv_b = (const float*)d_in[13];
  const float* bnkv_m = (const float*)d_in[14];
  const float* bnkv_v = (const float*)d_in[15];
  const float* bnds_g = (const float*)d_in[16];
  const float* bnds_b = (const float*)d_in[17];
  const float* bnds_m = (const float*)d_in[18];
  const float* bnds_v = (const float*)d_in[19];

  char* ws = (char*)d_ws;
  size_t off = 0;
  auto alloc = [&](size_t bytes) -> void* {
    void* p = ws + off;
    off += (bytes + 255) & ~(size_t)255;
    return p;
  };

  __bf16* qact   = (__bf16*)alloc((size_t)64 * 784 * 640 * 2);
  __bf16* xact   = (__bf16*)alloc((size_t)64 * 784 * 640 * 2);
  __bf16* kvact  = (__bf16*)alloc((size_t)64 * 224 * 640 * 2);
  __bf16* wq_bf  = (__bf16*)alloc((size_t)512 * 640 * 2);
  __bf16* wkv_bf = (__bf16*)alloc((size_t)1024 * 640 * 2);
  __bf16* wds_bf = (__bf16*)alloc((size_t)64 * 640 * 2);
  float*  beta   = (float*)alloc(64 * 4);
  __bf16* Qb     = (__bf16*)alloc((size_t)64 * 784 * 512 * 2);
  __bf16* Kbuf   = (__bf16*)alloc((size_t)64 * 224 * 512 * 2);
  __bf16* Vtb    = (__bf16*)alloc((size_t)64 * 512 * 224 * 2);
  float*  Rb     = (float*)alloc((size_t)64 * 784 * 64 * 4);
  float*  part   = (float*)alloc((size_t)64 * 8 * 49 * 64 * 4);

  // stage 0: weight conversions (BN of residual path folded into w_ds)
  cvt_bf16<<<dim3((512 * 640 + 255) / 256), 256, 0, stream>>>(w_pwq, wq_bf, 512 * 640);
  cvt_bf16<<<dim3((1024 * 640 + 255) / 256), 256, 0, stream>>>(w_pwkv, wkv_bf, 1024 * 640);
  cvt_wds<<<dim3((64 * 640 + 255) / 256), 256, 0, stream>>>(w_ds, bnds_g, bnds_b, bnds_m,
                                                            bnds_v, wds_bf, beta);

  // stage 1: depthwise conv + BN -> bf16 activations (K contiguous)
  stage1_q<<<dim3((784 * 640 + 255) / 256, 64), 256, 0, stream>>>(
      x, w_dwq, bnq_g, bnq_b, bnq_m, bnq_v, qact, xact);
  stage1_kv<<<dim3((224 * 640 + 255) / 256, 64), 256, 0, stream>>>(
      x, w_dwkv, bnkv_g, bnkv_b, bnkv_m, bnkv_v, kvact);

  // stage 2: WMMA GEMMs
  gemm_bf16<<<dim3(49, 8, 64), 32, 0, stream>>>(qact, wq_bf, Qb, nullptr,
      640, 784LL * 640, 784LL * 512, 512, 0);
  gemm_bf16<<<dim3(14, 8, 64), 32, 0, stream>>>(kvact, wkv_bf, Kbuf, nullptr,
      640, 224LL * 640, 224LL * 512, 512, 0);
  gemm_bf16<<<dim3(14, 8, 64), 32, 0, stream>>>(kvact, wkv_bf + (size_t)512 * 640, Vtb,
      nullptr, 640, 224LL * 640, 512LL * 224, 224, 1);
  gemm_bf16<<<dim3(49, 1, 64), 32, 0, stream>>>(xact, wds_bf, Rb, beta,
      640, 784LL * 640, 784LL * 64, 64, 2);

  // stage 3: fused attention (TDM async V prefetch) + residual + LN + partial pool
  attn_fused<<<dim3(49, 8, 64), 32, 0, stream>>>(Qb, Kbuf, Vtb, Rb, ln_g, ln_b, part);

  // stage 4: deterministic final pooling
  pool_final<<<dim3((64 * 8 * 64 + 255) / 256), 256, 0, stream>>>(part, (float*)d_out);
}
